// DCGRUCell_38577396252953
// MI455X (gfx1250) — matrix operand order repaired and checked by reading, compiled, not run on hardware
//
#include <hip/hip_runtime.h>

// ---------------------------------------------------------------------------
// Types for WMMA / async copies
// ---------------------------------------------------------------------------
typedef __attribute__((ext_vector_type(16))) __bf16          v16bf;
typedef __attribute__((ext_vector_type(8)))  float           v8f;
typedef __attribute__((ext_vector_type(8)))  unsigned short  v8us;
typedef __attribute__((ext_vector_type(16))) unsigned short  v16us;
typedef __attribute__((ext_vector_type(4)))  int             v4i;

__device__ __forceinline__ unsigned short f2bf(float f) {
  unsigned int u = __builtin_bit_cast(unsigned int, f);
  unsigned int r = u + 0x7FFFu + ((u >> 16) & 1u);   // round-to-nearest-even
  return (unsigned short)(r >> 16);
}
__device__ __forceinline__ float bf2f(unsigned short h) {
  unsigned int u = ((unsigned int)h) << 16;
  return __builtin_bit_cast(float, u);
}

// Scheduler hints: pin DS_READ / WMMA interleaving (no-op if unsupported)
#if __has_builtin(__builtin_amdgcn_sched_group_barrier)
#define SGB(mask, size, id) __builtin_amdgcn_sched_group_barrier((mask), (size), (id))
#else
#define SGB(mask, size, id)
#endif

// CDNA5 async global->LDS copies (ASYNCcnt-tracked), if the toolchain has them
#if __has_builtin(__builtin_amdgcn_global_load_async_to_lds_b128) && \
    __has_builtin(__builtin_amdgcn_s_wait_asynccnt)
#define HAVE_ASYNC_LDS 1
#else
#define HAVE_ASYNC_LDS 0
#endif

// ---------------------------------------------------------------------------
// Problem constants
// ---------------------------------------------------------------------------
#define NB    16          // batch
#define NN    4096        // nodes
#define NH    64          // hidden
#define NC    66          // Din + H = 2 + 64
#define NCOLS 1056        // NB * NC
#define KPAD  384         // 5*66 = 330 padded to multiple of BK=64 (no K guards)
#define KREAL 330

// GEMM tiling: 512 threads = 16 waves; wave (wrow,wcol) computes 16x64 strip
#define BM 128
#define BN 128
#define BKT 64
#define LDSS 72                 // padded LDS row stride (elements)
#define TILE_US (BM * LDSS)     // 9216 ushorts = 18 KB

// ---------------------------------------------------------------------------
// f32 -> bf16 bulk convert (8 elements / thread)
// ---------------------------------------------------------------------------
__global__ __launch_bounds__(256) void cvt_bf16x8(const float* __restrict__ src,
                                                  unsigned short* __restrict__ dst,
                                                  long n8) {
  long i = (long)blockIdx.x * 256 + threadIdx.x;
  if (i >= n8) return;
  const float4* s = (const float4*)src + i * 2;
  float4 a = s[0], b = s[1];
  uint4 o;
  o.x = (unsigned)f2bf(a.x) | ((unsigned)f2bf(a.y) << 16);
  o.y = (unsigned)f2bf(a.z) | ((unsigned)f2bf(a.w) << 16);
  o.z = (unsigned)f2bf(b.x) | ((unsigned)f2bf(b.y) << 16);
  o.w = (unsigned)f2bf(b.z) | ((unsigned)f2bf(b.w) << 16);
  *(uint4*)(dst + i * 8) = o;
}

// ---------------------------------------------------------------------------
// Weight convert: W_gate (330x128) -> Wgt[o][k] bf16 (128x384, zero padded)
//                 W_update (330x64) -> Wut[o][k] bf16 (128x384, zero padded)
// ---------------------------------------------------------------------------
__global__ __launch_bounds__(256) void cvt_w(const float* __restrict__ Wg,
                                             const float* __restrict__ Wu,
                                             unsigned short* __restrict__ Wgt,
                                             unsigned short* __restrict__ Wut) {
  int idx = blockIdx.x * 256 + threadIdx.x;
  if (idx >= 128 * KPAD) return;
  int o = idx / KPAD, k = idx % KPAD;
  Wgt[idx] = (k < KREAL) ? f2bf(Wg[(size_t)k * 128 + o]) : (unsigned short)0;
  Wut[idx] = (k < KREAL && o < NH) ? f2bf(Wu[(size_t)k * NH + o]) : (unsigned short)0;
}

// ---------------------------------------------------------------------------
// Build transposed feature matrices: xs_t[col][node], cols = b*66 + c
//   c<2: x,  c>=2: state.  Also writes the x columns of cand_t.
// ---------------------------------------------------------------------------
__global__ __launch_bounds__(256) void build_xt(const float* __restrict__ x,
                                                const float* __restrict__ state,
                                                unsigned short* __restrict__ xst,
                                                unsigned short* __restrict__ cand) {
  int b = blockIdx.y, n0 = blockIdx.x * 128, tid = threadIdx.x;
  __shared__ float st[128 * 65];
  for (int idx = tid; idx < 128 * 64; idx += 256) {
    int nl = idx >> 6, h = idx & 63;
    st[nl * 65 + h] = state[((size_t)b * NN + n0 + nl) * NH + h];
  }
  __syncthreads();
  for (int idx = tid; idx < 64 * 128; idx += 256) {
    int h = idx >> 7, nl = idx & 127;
    xst[((size_t)(b * NC + 2 + h)) * NN + n0 + nl] = f2bf(st[nl * 65 + h]);
  }
  for (int idx = tid; idx < 2 * 128; idx += 256) {
    int nl = idx >> 1, c = idx & 1;
    unsigned short v = f2bf(x[((size_t)b * NN + n0 + nl) * 2 + c]);
    xst [((size_t)(b * NC + c)) * NN + n0 + nl] = v;
    cand[((size_t)(b * NC + c)) * NN + n0 + nl] = v;
  }
}

// ---------------------------------------------------------------------------
// Pack H matrix: H[m][k], m = b*4096+n, k = f*66+c (zero-padded to 384)
// feats are transposed layout feat[col][node].  64-node tiles (50 KB LDS).
// ---------------------------------------------------------------------------
__global__ __launch_bounds__(256) void pack_h(const unsigned short* __restrict__ f0,
                                              const unsigned short* __restrict__ f1,
                                              const unsigned short* __restrict__ f2,
                                              const unsigned short* __restrict__ f3,
                                              const unsigned short* __restrict__ f4,
                                              unsigned short* __restrict__ H) {
  __shared__ __align__(16) unsigned short tile[64 * 392];  // 50176 B
  int b = blockIdx.y, n0 = blockIdx.x * 64, tid = threadIdx.x;
  const unsigned short* fs[5] = {f0, f1, f2, f3, f4};
  for (int idx = tid; idx < KREAL * 64; idx += 256) {
    int k = idx >> 6, nl = idx & 63;
    int f = k / NC, c = k % NC;
    tile[nl * 392 + k] = fs[f][(size_t)(b * NC + c) * NN + n0 + nl];
  }
  for (int idx = tid; idx < (KPAD - KREAL) * 64; idx += 256) {
    int k = KREAL + (idx >> 6), nl = idx & 63;
    tile[nl * 392 + k] = 0;
  }
  __syncthreads();
  for (int idx = tid; idx < 64 * (KPAD / 8); idx += 256) {
    int nl = idx / (KPAD / 8), q = idx % (KPAD / 8);
    *(uint4*)(H + ((size_t)(b * NN + n0 + nl)) * KPAD + q * 8) =
        *(const uint4*)(tile + nl * 392 + q * 8);
  }
}

// ---------------------------------------------------------------------------
// LDS fragment loader (two 16B chunks separated by d elements)
// ---------------------------------------------------------------------------
__device__ __forceinline__ v16bf ld_frag(const unsigned short* p, int d) {
  v8us lo = *(const v8us*)p;
  v8us hi = *(const v8us*)(p + d);
  v16us c;
#pragma unroll
  for (int q = 0; q < 8; ++q) { c[q] = lo[q]; c[8 + q] = hi[q]; }
  return __builtin_bit_cast(v16bf, c);
}

// ---------------------------------------------------------------------------
// Main bf16 GEMM with fused epilogues.  512 threads / 16 waves per block.
//   C[m][j] = sum_k Ag[m][k] * Bg[j][k]      (Bg stored "transposed": j-major)
// MODE 0 (diffusion): Yt[j][m] = bf16( scale*C - X0t[j][m] )
// MODE 1 (gate):      s = sigmoid(C + bias); o<64: cand = s*state; o>=64: rbuf = s
// MODE 2 (update):    hc = tanh(C + bias); out = r*state + (1-r)*hc  (o<64)
//
// Tile kt+1 is staged global->LDS at the TOP of iteration kt into the inactive
// buffer via CDNA5 async-to-LDS copies (no VGPR staging, ASYNCcnt-tracked);
// fallback is an adjacent load->store pair with few-instruction live ranges.
// ---------------------------------------------------------------------------
template <int MODE>
__global__ __launch_bounds__(512) void gemm_bf16(
    const unsigned short* __restrict__ Ag,   // M x K row-major
    const unsigned short* __restrict__ Bg,   // Ncols x K row-major (B^T storage)
    int M, int Ncols, int K, int lda, int ldb,
    unsigned short* __restrict__ Yt,         // MODE0: Ncols x M
    const unsigned short* __restrict__ X0t,  // MODE0 optional
    float scale,
    const float* __restrict__ bias,
    const float* __restrict__ state,
    unsigned short* __restrict__ cand,
    float* __restrict__ rbuf,
    float* __restrict__ outp) {
  __shared__ __align__(16) unsigned short smem[4 * TILE_US];  // A0 A1 B0 B1 / Cst alias

  int tid = threadIdx.x;
  int wave = tid >> 5, lane = tid & 31;
  int lanelo = lane & 15, hi = lane >> 4;
  int wrow = wave >> 1, wcol = wave & 1;      // 8 row strips x 2 col halves
  int row0 = blockIdx.y * BM, col0 = blockIdx.x * BN;

  v8f acc[4];
#pragma unroll
  for (int t = 0; t < 4; ++t)
#pragma unroll
    for (int i = 0; i < 8; ++i) acc[t][i] = 0.f;

  // Per-thread global/LDS element offsets (32-bit; max ~25M elements)
  unsigned aoff[2], boff[2];
  int soff[2];
#pragma unroll
  for (int j = 0; j < 2; ++j) {
    int idx = tid + j * 512;
    int rr = idx >> 3, kc = (idx & 7) << 3;
    aoff[j] = (unsigned)(row0 + rr) * (unsigned)lda + (unsigned)kc;
    int bc = col0 + rr;
    bc = (bc > Ncols - 1) ? (Ncols - 1) : bc;       // clamp -> branch-free loads
    boff[j] = (unsigned)bc * (unsigned)ldb + (unsigned)kc;
    soff[j] = rr * LDSS + kc;
  }

  // Stage one 128x64(A) + 128x64(B) tile pair into LDS buffers (nb, 2+nb)
#if HAVE_ASYNC_LDS
#define STAGE_TILE(K0N, NBUF)                                                   \
  {                                                                             \
    _Pragma("unroll")                                                           \
    for (int j_ = 0; j_ < 2; ++j_) {                                            \
      __builtin_amdgcn_global_load_async_to_lds_b128(                           \
          (__attribute__((address_space(1))) v4i*)(Ag + aoff[j_] + (K0N)),      \
          (__attribute__((address_space(3))) v4i*)(smem + (NBUF) * TILE_US + soff[j_]), 0, 0); \
      __builtin_amdgcn_global_load_async_to_lds_b128(                           \
          (__attribute__((address_space(1))) v4i*)(Bg + boff[j_] + (K0N)),      \
          (__attribute__((address_space(3))) v4i*)(smem + (2 + (NBUF)) * TILE_US + soff[j_]), 0, 0); \
    }                                                                           \
  }
#define STAGE_WAIT() __builtin_amdgcn_s_wait_asynccnt(0)
#else
#define STAGE_TILE(K0N, NBUF)                                                   \
  {                                                                             \
    uint4 ra_[2], rb_[2];                                                       \
    _Pragma("unroll")                                                           \
    for (int j_ = 0; j_ < 2; ++j_) {                                            \
      ra_[j_] = *(const uint4*)(Ag + aoff[j_] + (K0N));                         \
      rb_[j_] = *(const uint4*)(Bg + boff[j_] + (K0N));                         \
    }                                                                           \
    _Pragma("unroll")                                                           \
    for (int j_ = 0; j_ < 2; ++j_) {                                            \
      *(uint4*)(smem + (NBUF) * TILE_US + soff[j_]) = ra_[j_];                  \
      *(uint4*)(smem + (2 + (NBUF)) * TILE_US + soff[j_]) = rb_[j_];            \
    }                                                                           \
  }
#define STAGE_WAIT()
#endif

  STAGE_TILE(0u, 0);
  STAGE_WAIT();
  __syncthreads();

  int abase = (wrow * 16 + lanelo) * LDSS + hi * 8;
  int bbase = (wcol * 64 + lanelo) * LDSS + hi * 8;

  int NK = K / BKT;
  for (int kt = 0; kt < NK; ++kt) {
    int cur = kt & 1;
    bool more = (kt + 1) < NK;
    if (more) {
      // stage tile kt+1 into the inactive buffer while computing tile kt
      unsigned k0n = (unsigned)(kt + 1) * BKT;
      STAGE_TILE(k0n, (kt + 1) & 1);
    }

    const unsigned short* Ac = smem + cur * TILE_US;
    const unsigned short* Bc = smem + (2 + cur) * TILE_US;
#pragma unroll
    for (int ks = 0; ks < BKT; ks += 32) {
      v16bf av = ld_frag(Ac + abase + ks, 16);
      // one-fragment lookahead: load B(t+1) before WMMA(t)
      v16bf bcur = ld_frag(Bc + bbase + ks, 16);
#pragma unroll
      for (int t = 0; t < 4; ++t) {
        v16bf bnxt = bcur;
        if (t < 3) bnxt = ld_frag(Bc + bbase + ks + (t + 1) * (16 * LDSS), 16);
        acc[t] = __builtin_amdgcn_wmma_f32_16x16x32_bf16(
            false, av, false, bcur, (short)0, acc[t], false, false);
        bcur = bnxt;
      }
      // Pin schedule: [A+B0+B1 : 6 ds_read][wmma][2 ds_read][wmma][2 ds_read][wmma][wmma]
      SGB(0x100, 6, 0);
      SGB(0x008, 1, 0);
      SGB(0x100, 2, 0);
      SGB(0x008, 1, 0);
      SGB(0x100, 2, 0);
      SGB(0x008, 1, 0);
      SGB(0x008, 1, 0);
    }

    if (more) STAGE_WAIT();
    __syncthreads();
  }

  if constexpr (MODE == 0) {
    // Stage C through LDS, store transposed bf16 (coalesced along nodes)
    float* Cst = (float*)smem;  // 128 x 132 f32 = 67584 B <= 73728 B
#pragma unroll
    for (int t = 0; t < 4; ++t)
#pragma unroll
      for (int i = 0; i < 8; ++i)
        Cst[(wrow * 16 + hi * 8 + i) * 132 + wcol * 64 + t * 16 + lanelo] = acc[t][i];
    __syncthreads();

    int j = tid >> 2, quarter = tid & 3;        // 128 cols x 4 row-chunks of 32
    int jg = col0 + j;
    if (jg < Ncols) {
      const unsigned short* x0p =
          X0t ? (X0t + (size_t)jg * M + row0 + quarter * 32) : (const unsigned short*)0;
      uint4* dst = (uint4*)(Yt + (size_t)jg * M + row0 + quarter * 32);
#pragma unroll
      for (int q = 0; q < 4; ++q) {
        unsigned pk[4];
#pragma unroll
        for (int p = 0; p < 4; ++p) {
          int m = q * 8 + p * 2;
          float v0 = scale * Cst[(quarter * 32 + m) * 132 + j];
          float v1 = scale * Cst[(quarter * 32 + m + 1) * 132 + j];
          if (X0t) { v0 -= bf2f(x0p[m]); v1 -= bf2f(x0p[m + 1]); }
          pk[p] = (unsigned)f2bf(v0) | ((unsigned)f2bf(v1) << 16);
        }
        uint4 v; v.x = pk[0]; v.y = pk[1]; v.z = pk[2]; v.w = pk[3];
        dst[q] = v;
      }
    }
  } else {
#pragma unroll
    for (int t = 0; t < 4; ++t) {
      int o = col0 + wcol * 64 + t * 16 + lanelo;
#pragma unroll
      for (int i = 0; i < 8; ++i) {
        int m = row0 + wrow * 16 + hi * 8 + i;  // m = b*4096 + n
        if constexpr (MODE == 1) {
          float val = acc[t][i] + bias[o];
          float s = 1.f / (1.f + __expf(-val));
          int b = m >> 12, n = m & 4095;
          if (o < NH) {
            cand[(size_t)(b * NC + 2 + o) * NN + n] = f2bf(s * state[(size_t)m * NH + o]);
          } else {
            rbuf[(size_t)m * NH + (o - NH)] = s;
          }
        } else {
          if (o < NH) {
            float val = acc[t][i] + bias[o];
            float hc = tanhf(val);
            float r = rbuf[(size_t)m * NH + o];
            float st = state[(size_t)m * NH + o];
            outp[(size_t)m * NH + o] = r * st + (1.f - r) * hc;
          }
        }
      }
    }
  }
}

// ---------------------------------------------------------------------------
// Launcher
// ---------------------------------------------------------------------------
extern "C" void kernel_launch(void* const* d_in, const int* in_sizes, int n_in,
                              void* d_out, int out_size, void* d_ws, size_t ws_size,
                              hipStream_t stream) {
  (void)in_sizes; (void)n_in; (void)out_size; (void)ws_size;
  const float* x     = (const float*)d_in[0];
  const float* state = (const float*)d_in[1];
  const float* sup0  = (const float*)d_in[2];
  const float* sup1  = (const float*)d_in[3];
  const float* Wg    = (const float*)d_in[4];
  const float* bg    = (const float*)d_in[5];
  const float* Wu    = (const float*)d_in[6];
  const float* bu    = (const float*)d_in[7];
  float* out = (float*)d_out;

  char* w = (char*)d_ws;
  auto take = [&](size_t bytes) {
    char* p = w;
    w += (bytes + 255) & ~(size_t)255;
    return p;
  };
  unsigned short* A0b  = (unsigned short*)take((size_t)NN * NN * 2);        // 32 MB
  unsigned short* A1b  = (unsigned short*)take((size_t)NN * NN * 2);        // 32 MB
  unsigned short* xst  = (unsigned short*)take((size_t)NCOLS * NN * 2);
  unsigned short* cand = (unsigned short*)take((size_t)NCOLS * NN * 2);
  unsigned short* F1   = (unsigned short*)take((size_t)NCOLS * NN * 2);
  unsigned short* F2   = (unsigned short*)take((size_t)NCOLS * NN * 2);
  unsigned short* F3   = (unsigned short*)take((size_t)NCOLS * NN * 2);
  unsigned short* F4   = (unsigned short*)take((size_t)NCOLS * NN * 2);
  unsigned short* Hb   = (unsigned short*)take((size_t)NB * NN * KPAD * 2); // 48 MB
  float*          rb   = (float*)take((size_t)NB * NN * NH * 4);            // 16 MB
  unsigned short* Wgt  = (unsigned short*)take((size_t)128 * KPAD * 2);
  unsigned short* Wut  = (unsigned short*)take((size_t)128 * KPAD * 2);

  dim3 blk(256), blkg(512);
  // Precision conversion (supports stay resident in L2 in bf16: 64 MB total)
  cvt_bf16x8<<<8192, blk, 0, stream>>>(sup0, A0b, (long)NN * NN / 8);
  cvt_bf16x8<<<8192, blk, 0, stream>>>(sup1, A1b, (long)NN * NN / 8);
  cvt_w<<<(128 * KPAD + 255) / 256, blk, 0, stream>>>(Wg, Wu, Wgt, Wut);
  build_xt<<<dim3(NN / 128, NB), blk, 0, stream>>>(x, state, xst, cand);

  dim3 gdiff((NCOLS + BN - 1) / BN, NN / BM);     // 9 x 32
  dim3 gproj(1, (NB * NN) / BM);                  // 1 x 512

  // ---- gate GCN diffusion: f1 = A0 xs ; f2 = 2 A0 f1 - xs ; f3 = A1 xs ; f4 = 2 A1 f3 - xs
  gemm_bf16<0><<<gdiff, blkg, 0, stream>>>(A0b, xst, NN, NCOLS, NN, NN, NN, F1, nullptr, 1.f,
                                           nullptr, nullptr, nullptr, nullptr, nullptr);
  gemm_bf16<0><<<gdiff, blkg, 0, stream>>>(A0b, F1, NN, NCOLS, NN, NN, NN, F2, xst, 2.f,
                                           nullptr, nullptr, nullptr, nullptr, nullptr);
  gemm_bf16<0><<<gdiff, blkg, 0, stream>>>(A1b, xst, NN, NCOLS, NN, NN, NN, F3, nullptr, 1.f,
                                           nullptr, nullptr, nullptr, nullptr, nullptr);
  gemm_bf16<0><<<gdiff, blkg, 0, stream>>>(A1b, F3, NN, NCOLS, NN, NN, NN, F4, xst, 2.f,
                                           nullptr, nullptr, nullptr, nullptr, nullptr);
  pack_h<<<dim3(NN / 64, NB), blk, 0, stream>>>(xst, F1, F2, F3, F4, Hb);
  // gate projection + sigmoid; writes z*state into cand cols, r into rb
  gemm_bf16<1><<<gproj, blkg, 0, stream>>>(Hb, Wgt, NB * NN, 128, KPAD, KPAD, KPAD,
                                           nullptr, nullptr, 1.f, bg, state, cand, rb, nullptr);

  // ---- candidate GCN diffusion on cand_in
  gemm_bf16<0><<<gdiff, blkg, 0, stream>>>(A0b, cand, NN, NCOLS, NN, NN, NN, F1, nullptr, 1.f,
                                           nullptr, nullptr, nullptr, nullptr, nullptr);
  gemm_bf16<0><<<gdiff, blkg, 0, stream>>>(A0b, F1, NN, NCOLS, NN, NN, NN, F2, cand, 2.f,
                                           nullptr, nullptr, nullptr, nullptr, nullptr);
  gemm_bf16<0><<<gdiff, blkg, 0, stream>>>(A1b, cand, NN, NCOLS, NN, NN, NN, F3, nullptr, 1.f,
                                           nullptr, nullptr, nullptr, nullptr, nullptr);
  gemm_bf16<0><<<gdiff, blkg, 0, stream>>>(A1b, F3, NN, NCOLS, NN, NN, NN, F4, cand, 2.f,
                                           nullptr, nullptr, nullptr, nullptr, nullptr);
  pack_h<<<dim3(NN / 64, NB), blk, 0, stream>>>(cand, F1, F2, F3, F4, Hb);
  // update projection + tanh + final GRU combine -> out (f32, (B,N,H))
  gemm_bf16<2><<<gproj, blkg, 0, stream>>>(Hb, Wut, NB * NN, 128, KPAD, KPAD, KPAD,
                                           nullptr, nullptr, 1.f, bu, state, nullptr, rb, out);
}
